// Attention_39221641347258
// MI455X (gfx1250) — matrix operand order repaired and checked by reading, not compiled, not run on hardware
//
#include <hip/hip_runtime.h>
#include <hip/hip_bf16.h>

// ---------------------------------------------------------------------------
// Transformer-XL attention block for MI455X (gfx1250).
// bf16 WMMA pipeline + Tensor Data Mover (TDM) tile DMA where available.
// ---------------------------------------------------------------------------

typedef __attribute__((ext_vector_type(16))) __bf16       v16bf;
typedef __attribute__((ext_vector_type(8)))  float        v8f;
typedef __attribute__((ext_vector_type(4)))  unsigned int v4u;
typedef __attribute__((ext_vector_type(8)))  int          v8i;
typedef __attribute__((ext_vector_type(4)))  int          v4i;

__device__ inline v8f wmma_bf16(v16bf a, v16bf b, v8f c) {
  return __builtin_amdgcn_wmma_f32_16x16x32_bf16(
      /*neg_a=*/false, a, /*neg_b=*/false, b,
      /*c_mod=*/(short)0, c, /*reuse_a=*/false, /*reuse_b=*/false);
}

// ---- TDM availability (device pass only; host pass parses the fallback) ----
#define XL_TDM 0
#if defined(__gfx1250__)
#  ifdef __has_builtin
#    if __has_builtin(__builtin_amdgcn_tensor_load_to_lds) && \
        __has_builtin(__builtin_amdgcn_s_wait_tensorcnt)
#      undef XL_TDM
#      define XL_TDM 1
#    endif
#  endif
#endif

#if XL_TDM
// 2D tile DMA: global (row-major, 2-byte elements) -> LDS, with LDS row padding.
// D# layout per CDNA5 ISA (groups 0/1; groups 2/3 zero
// for <=2D tensors). pad_interval: DWORDs before pad = 2^(n+1); pad_amount:
// pad DWORDs = n+1.
__device__ inline void tdm_load_2d(unsigned lds_byte_off, const void* gptr,
                                   unsigned tensor_d0, unsigned tensor_d1,
                                   unsigned tile_d0, unsigned tile_d1,
                                   unsigned stride0,
                                   unsigned pad_interval, unsigned pad_amount) {
  unsigned long long ga = (unsigned long long)gptr;
  v4u g0;
  g0[0] = 1u;                                            // count=1, user desc
  g0[1] = lds_byte_off;                                  // lds_addr
  g0[2] = (unsigned)(ga & 0xffffffffu);                  // global_addr[31:0]
  g0[3] = (unsigned)((ga >> 32) & 0x01ffffffu) | (2u << 30);  // addr[56:32] | type=2
  v8i g1;
  g1[0] = (int)((1u << 16)                               // data_size = 2 bytes
              | (1u << 20)                               // pad_enable
              | (pad_interval << 22) | (pad_amount << 25));
  g1[1] = (int)((tensor_d0 & 0xffffu) << 16);            // barrier addr=0 | td0.lo
  g1[2] = (int)((tensor_d0 >> 16) | ((tensor_d1 & 0xffffu) << 16));
  g1[3] = (int)((tensor_d1 >> 16) | (tile_d0 << 16));
  g1[4] = (int)(tile_d1 & 0xffffu);                      // tile_dim2 = 0
  g1[5] = (int)stride0;                                  // dim0 stride (lo32)
  g1[6] = 0;                                             // stride0.hi | stride1.lo
  g1[7] = 0;
  v4i z4 = {};
#if __clang_major__ >= 23
  v8i z8 = {};
  __builtin_amdgcn_tensor_load_to_lds(g0, g1, z4, z4, z8, 0);
#else
  __builtin_amdgcn_tensor_load_to_lds(g0, g1, z4, z4, 0);
#endif
}
__device__ inline unsigned lds_off32(const void* p) {
  return (unsigned)(unsigned long long)p;   // low 32 bits of generic LDS addr
}
#endif  // XL_TDM

// ---------------------------------------------------------------------------
// fp32 -> bf16 cast
// ---------------------------------------------------------------------------
__global__ __launch_bounds__(256)
void xl_cast_bf16(const float* __restrict__ in, __bf16* __restrict__ out, size_t n) {
  size_t i = (size_t)blockIdx.x * blockDim.x + threadIdx.x;
  size_t stride = (size_t)gridDim.x * blockDim.x;
  for (; i < n; i += stride) out[i] = (__bf16)in[i];
}

// ---------------------------------------------------------------------------
// bf16 GEMM: C[M,N] = A[M,K]*B[K,N]; 128x128 tile, K-step 32, 8 waves.
// MODE 0: out0 = bf16(acc + bias0[col&63]); out1 = bf16(acc + bias1[col&63])
// MODE 1: out0 = bf16(acc)
// MODE 2: outF = acc + residual
// A tile via TDM DMA (pad matches LDS row pad); B tile transposed manually.
// ---------------------------------------------------------------------------
#define GP 8  // LDS pad (halves) == 4 DWORDs

template <int MODE>
__global__ __launch_bounds__(256)
void xl_gemm_bf16(const __bf16* __restrict__ A, const __bf16* __restrict__ Bw,
                  int M, int N, int Kd,
                  float* __restrict__ outF, const float* __restrict__ residual,
                  __bf16* __restrict__ out0, const float* __restrict__ bias0,
                  __bf16* __restrict__ out1, const float* __restrict__ bias1) {
  __shared__ __bf16 At[128][32 + GP];
  __shared__ __bf16 Bt[128][32 + GP];  // transposed: [n][k]

  const int tid  = threadIdx.x;
  const int lane = tid & 31;
  const int wave = tid >> 5;
  const int m0 = blockIdx.y * 128;
  const int n0 = blockIdx.x * 128;
  const int wm = (wave & 3) * 32;
  const int wn = (wave >> 2) * 64;

#if XL_TDM
  const unsigned at_off = lds_off32(&At[0][0]);
#endif

  v8f acc[2][4] = {};

  for (int k0 = 0; k0 < Kd; k0 += 32) {
    __syncthreads();
#if XL_TDM
    // A tile: 128 rows x 32 halves, DMA'd with 16-DWORD interval / 4-DWORD pad
    if (wave == 0)
      tdm_load_2d(at_off, A + (size_t)m0 * Kd + k0,
                  (unsigned)Kd, (unsigned)M, 32u, 128u, (unsigned)Kd,
                  /*interval 2^(3+1)=16 DW*/ 3u, /*pad 4 DW*/ 3u);
#else
    {
      int row = tid >> 1, col = (tid & 1) * 16;
      const __bf16* src = A + (size_t)(m0 + row) * Kd + k0 + col;
#pragma unroll
      for (int j = 0; j < 16; ++j) At[row][col + j] = src[j];
    }
#endif
    {  // B tile transposed: Bw[k][n] -> Bt[n][k]
      int kk = tid >> 3, nb = (tid & 7) * 16;
      const __bf16* src = Bw + (size_t)(k0 + kk) * N + n0 + nb;
      __builtin_prefetch(src + 32 * (size_t)N, 0, 1);  // next k-slab
#pragma unroll
      for (int j = 0; j < 16; ++j) Bt[nb + j][kk] = src[j];
    }
#if XL_TDM
    if (wave == 0) __builtin_amdgcn_s_wait_tensorcnt((short)0);
#endif
    __syncthreads();

    v16bf af[2];
#pragma unroll
    for (int r = 0; r < 2; ++r) {
      int m  = wm + r * 16 + (lane & 15);
      int b8 = (lane < 16) ? 0 : 16;
#pragma unroll
      for (int j = 0; j < 8; ++j) {
        af[r][j]     = At[m][b8 + j];
        af[r][8 + j] = At[m][b8 + 8 + j];
      }
    }
#pragma unroll
    for (int c = 0; c < 4; ++c) {
      v16bf bf;
      int n  = wn + c * 16 + (lane & 15);
      int kh = (lane < 16) ? 0 : 16;
#pragma unroll
      for (int j = 0; j < 16; ++j) bf[j] = Bt[n][kh + j];
#pragma unroll
      for (int r = 0; r < 2; ++r) acc[r][c] = wmma_bf16(af[r], bf, acc[r][c]);
    }
  }

  // Epilogue (straight-line per MODE).
#pragma unroll
  for (int r = 0; r < 2; ++r)
#pragma unroll
    for (int c = 0; c < 4; ++c)
#pragma unroll
      for (int g = 0; g < 8; ++g) {
        int row = m0 + wm + r * 16 + g + ((lane >= 16) ? 8 : 0);
        int col = n0 + wn + c * 16 + (lane & 15);
        size_t idx = (size_t)row * N + col;
        float vv = acc[r][c][g];
        if (MODE == 0) {
          out0[idx] = (__bf16)(vv + bias0[col & 63]);
          out1[idx] = (__bf16)(vv + bias1[col & 63]);
        } else if (MODE == 1) {
          out0[idx] = (__bf16)vv;
        } else {
          outF[idx] = vv + residual[idx];
        }
      }
}

// ---------------------------------------------------------------------------
// Flash-style Transformer-XL attention.
// grid = (B*H, TQ/64), block = 128 (4 waves x 16 q-rows).
// S = ((q+u)K^T + (q+v)R^T)/8, causal j <= i+mem, online softmax, ctx += P*V.
// Qu/Qv/K/R tiles DMA'd by TDM; V transposed manually (TDM can't transpose).
// ---------------------------------------------------------------------------
#define TQB 64
#define KC  32
#define HDV 64
#define AP  8

__global__ __launch_bounds__(128)
void xl_attention(const __bf16* __restrict__ qu, const __bf16* __restrict__ qv,
                  const __bf16* __restrict__ kp, const __bf16* __restrict__ rp,
                  const __bf16* __restrict__ vp, __bf16* __restrict__ ctxb,
                  int Bn, int TQn, int TKn, int Hn) {
  __shared__ __bf16 sQu[TQB][HDV + AP];
  __shared__ __bf16 sQv[TQB][HDV + AP];
  __shared__ __bf16 sK[KC][HDV + AP];
  __shared__ __bf16 sR[KC][HDV + AP];
  __shared__ __bf16 sVt[HDV][KC + AP];   // V transposed: [dv][tk]
  __shared__ __bf16 sP[4][16][KC + AP];  // per-wave P tile

  const int tid  = threadIdx.x;
  const int lane = tid & 31;
  const int wave = tid >> 5;
  const int bh = blockIdx.x;
  const int b = bh / Hn, h = bh % Hn;
  const int q0 = blockIdx.y * TQB;
  const int DPn = Hn * HDV;
  const int mem = TKn - TQn;

  const size_t qbase = ((size_t)b * TQn + q0) * DPn + h * HDV;
  const size_t kbase = ((size_t)b * TKn) * DPn + h * HDV;

  // ---- load q-tiles (64 x 64) once ----
#if XL_TDM
  if (wave == 0)
    tdm_load_2d(lds_off32(&sQu[0][0]), qu + qbase,
                (unsigned)DPn, (unsigned)TQn, (unsigned)HDV, (unsigned)TQB,
                (unsigned)DPn, /*32 DW*/ 4u, /*4 DW*/ 3u);
  else if (wave == 1)
    tdm_load_2d(lds_off32(&sQv[0][0]), qv + qbase,
                (unsigned)DPn, (unsigned)TQn, (unsigned)HDV, (unsigned)TQB,
                (unsigned)DPn, 4u, 3u);
  if (wave < 2) __builtin_amdgcn_s_wait_tensorcnt((short)0);
#else
  for (int it = tid; it < TQB * 4; it += 128) {
    int row = it >> 2, cg = (it & 3) * 16;
    const __bf16* su = qu + qbase + (size_t)row * DPn + cg;
    const __bf16* sv = qv + qbase + (size_t)row * DPn + cg;
#pragma unroll
    for (int j = 0; j < 16; ++j) { sQu[row][cg + j] = su[j]; sQv[row][cg + j] = sv[j]; }
  }
#endif
  __syncthreads();

  float mrun[8], lrun[8];
  v8f ctx[4] = {};
#pragma unroll
  for (int g = 0; g < 8; ++g) { mrun[g] = -3.0e38f; lrun[g] = 0.0f; }

  const int kend = (TKn < q0 + TQB + mem) ? TKn : (q0 + TQB + mem);

  for (int kb = 0; kb < kend; kb += KC) {
    __syncthreads();
#if XL_TDM
    if (wave == 0)
      tdm_load_2d(lds_off32(&sK[0][0]), kp + kbase + (size_t)kb * DPn,
                  (unsigned)DPn, (unsigned)TKn, (unsigned)HDV, (unsigned)KC,
                  (unsigned)DPn, 4u, 3u);
    else if (wave == 1)
      tdm_load_2d(lds_off32(&sR[0][0]), rp + kbase + (size_t)kb * DPn,
                  (unsigned)DPn, (unsigned)TKn, (unsigned)HDV, (unsigned)KC,
                  (unsigned)DPn, 4u, 3u);
#endif
    {  // V^T tile (all threads; transpose, so no TDM)
      int row = tid >> 2, cg = (tid & 3) * 16;
      const __bf16* svv = vp + kbase + (size_t)(kb + row) * DPn + cg;
#pragma unroll
      for (int j = 0; j < 16; ++j) sVt[cg + j][row] = svv[j];
#if !XL_TDM
      const __bf16* skk = kp + kbase + (size_t)(kb + row) * DPn + cg;
      const __bf16* srr = rp + kbase + (size_t)(kb + row) * DPn + cg;
#pragma unroll
      for (int j = 0; j < 16; ++j) { sK[row][cg + j] = skk[j]; sR[row][cg + j] = srr[j]; }
#endif
    }
#if XL_TDM
    if (wave < 2) __builtin_amdgcn_s_wait_tensorcnt((short)0);
#endif
    __syncthreads();

    // ---- scores: 16 rows x 32 keys, contraction over dv=64 (2 slices) ----
    v8f c0 = {}, c1 = {};
#pragma unroll
    for (int s = 0; s < 2; ++s) {
      v16bf aqu, aqv;
      {
        int m  = wave * 16 + (lane & 15);
        int b8 = ((lane < 16) ? 0 : 16) + s * 32;
#pragma unroll
        for (int j = 0; j < 8; ++j) {
          aqu[j] = sQu[m][b8 + j];  aqu[8 + j] = sQu[m][b8 + 8 + j];
          aqv[j] = sQv[m][b8 + j];  aqv[8 + j] = sQv[m][b8 + 8 + j];
        }
      }
#pragma unroll
      for (int cc = 0; cc < 2; ++cc) {
        v16bf bk, br;
        int n  = cc * 16 + (lane & 15);
        int kh = ((lane < 16) ? 0 : 16) + s * 32;
#pragma unroll
        for (int j = 0; j < 16; ++j) { bk[j] = sK[n][kh + j]; br[j] = sR[n][kh + j]; }
        if (cc == 0) { c0 = wmma_bf16(aqu, bk, c0); c0 = wmma_bf16(aqv, br, c0); }
        else         { c1 = wmma_bf16(aqu, bk, c1); c1 = wmma_bf16(aqv, br, c1); }
      }
    }

    // ---- online softmax ----
    const int ibase = q0 + wave * 16 + ((lane >= 16) ? 8 : 0);
#pragma unroll
    for (int g = 0; g < 8; ++g) {
      int irow = ibase + g;
      int j0 = kb + (lane & 15);
      int j1 = j0 + 16;
      float s0 = c0[g] * 0.125f;
      float s1 = c1[g] * 0.125f;
      if (j0 > irow + mem) s0 = -3.0e38f;
      if (j1 > irow + mem) s1 = -3.0e38f;
      float mx = fmaxf(s0, s1);
#pragma unroll
      for (int d = 8; d >= 1; d >>= 1) mx = fmaxf(mx, __shfl_xor(mx, d, 32));
      float mn = fmaxf(mrun[g], mx);
      float p0 = __expf(s0 - mn), p1 = __expf(s1 - mn);
      float sum = p0 + p1;
#pragma unroll
      for (int d = 8; d >= 1; d >>= 1) sum += __shfl_xor(sum, d, 32);
      float alpha = __expf(mrun[g] - mn);
      lrun[g] = lrun[g] * alpha + sum;
      mrun[g] = mn;
      int prow = g + ((lane >= 16) ? 8 : 0);
      sP[wave][prow][lane & 15]        = (__bf16)p0;
      sP[wave][prow][16 + (lane & 15)] = (__bf16)p1;
#pragma unroll
      for (int t = 0; t < 4; ++t) ctx[t][g] *= alpha;
    }
    asm volatile("s_wait_dscnt 0x0" ::: "memory");  // wave-private P round-trip

    // ---- ctx += P * V ----
    v16bf ap;
    {
      int m  = lane & 15;
      int b8 = (lane < 16) ? 0 : 16;
#pragma unroll
      for (int j = 0; j < 8; ++j) {
        ap[j]     = sP[wave][m][b8 + j];
        ap[8 + j] = sP[wave][m][b8 + 8 + j];
      }
    }
#pragma unroll
    for (int t = 0; t < 4; ++t) {
      v16bf bv;
      int n  = t * 16 + (lane & 15);
      int kh = (lane < 16) ? 0 : 16;
#pragma unroll
      for (int j = 0; j < 16; ++j) bv[j] = sVt[n][kh + j];
      ctx[t] = wmma_bf16(ap, bv, ctx[t]);
    }
  }

  // ---- write ctx / l ----
  const size_t obase = ((size_t)b * TQn) * DPn + h * HDV;
#pragma unroll
  for (int t = 0; t < 4; ++t)
#pragma unroll
    for (int g = 0; g < 8; ++g) {
      int row = q0 + wave * 16 + g + ((lane >= 16) ? 8 : 0);
      int col = t * 16 + (lane & 15);
      ctxb[obase + (size_t)row * DPn + col] = (__bf16)(ctx[t][g] / lrun[g]);
    }
}

// ---------------------------------------------------------------------------
// LayerNorm: one block per row of D floats.
// ---------------------------------------------------------------------------
__global__ __launch_bounds__(256)
void xl_layernorm(const float* __restrict__ x, const float* __restrict__ gamma,
                  const float* __restrict__ beta, float* __restrict__ out, int Dn) {
  const int row = blockIdx.x;
  const float* xr = x + (size_t)row * Dn;
  const int tid = threadIdx.x, lane = tid & 31, wave = tid >> 5;
  __shared__ float rs[8], rs2[8];
  float s = 0.0f, s2 = 0.0f;
  for (int i = tid; i < Dn; i += 256) { float v = xr[i]; s += v; s2 += v * v; }
#pragma unroll
  for (int d = 16; d >= 1; d >>= 1) { s += __shfl_xor(s, d, 32); s2 += __shfl_xor(s2, d, 32); }
  if (lane == 0) { rs[wave] = s; rs2[wave] = s2; }
  __syncthreads();
  float ts = 0.0f, ts2 = 0.0f;
#pragma unroll
  for (int w = 0; w < 8; ++w) { ts += rs[w]; ts2 += rs2[w]; }
  float mu  = ts / Dn;
  float var = ts2 / Dn - mu * mu;
  float inv = rsqrtf(var + 1e-5f);
  for (int i = tid; i < Dn; i += 256)
    out[(size_t)row * Dn + i] = (xr[i] - mu) * inv * gamma[i] + beta[i];
}

// ---------------------------------------------------------------------------
// Host orchestration
// ---------------------------------------------------------------------------
extern "C" void kernel_launch(void* const* d_in, const int* in_sizes, int n_in,
                              void* d_out, int out_size, void* d_ws, size_t ws_size,
                              hipStream_t stream) {
  (void)in_sizes; (void)n_in; (void)out_size; (void)ws_size;
  enum { Bn = 4, TQn = 1024, TKn = 1536, Dn = 1024, Hn = 16, DVn = 64, DPn = 1024 };

  const float* query     = (const float*)d_in[0];
  const float* key_value = (const float*)d_in[1];
  const float* relative  = (const float*)d_in[2];
  /* d_in[3] = mask: Transformer-XL causal mask, recomputed analytically */
  const float* Wq = (const float*)d_in[4];
  const float* Wk = (const float*)d_in[5];
  const float* Wv = (const float*)d_in[6];
  const float* Wr = (const float*)d_in[7];
  const float* Wo = (const float*)d_in[8];
  const float* u  = (const float*)d_in[9];
  const float* v  = (const float*)d_in[10];
  const float* gm = (const float*)d_in[11];
  const float* bt = (const float*)d_in[12];
  float* out = (float*)d_out;

  const size_t nQ  = (size_t)Bn * TQn * Dn;
  const size_t nKV = (size_t)Bn * TKn * Dn;
  const size_t nW  = (size_t)Dn * DPn;

  char* p = (char*)d_ws;
  auto alloc = [&](size_t bytes) -> char* {
    char* r = p; p += (bytes + 255) & ~(size_t)255; return r;
  };
  __bf16* qbf   = (__bf16*)alloc(nQ * 2);
  __bf16* kvbf  = (__bf16*)alloc(nKV * 2);
  __bf16* rbf   = (__bf16*)alloc(nKV * 2);
  __bf16* wqbf  = (__bf16*)alloc(nW * 2);
  __bf16* wkbf  = (__bf16*)alloc(nW * 2);
  __bf16* wvbf  = (__bf16*)alloc(nW * 2);
  __bf16* wrbf  = (__bf16*)alloc(nW * 2);
  __bf16* wobf  = (__bf16*)alloc(nW * 2);
  __bf16* quB   = (__bf16*)alloc(nQ * 2);
  __bf16* qvB   = (__bf16*)alloc(nQ * 2);
  __bf16* kpB   = (__bf16*)alloc(nKV * 2);
  __bf16* vpB   = (__bf16*)alloc(nKV * 2);
  __bf16* rpB   = (__bf16*)alloc(nKV * 2);
  __bf16* ctxB  = (__bf16*)alloc(nQ * 2);
  float*  preln = (float*)alloc(nQ * 4);

  // 1) casts
  xl_cast_bf16<<<2048, 256, 0, stream>>>(query, qbf, nQ);
  xl_cast_bf16<<<2048, 256, 0, stream>>>(key_value, kvbf, nKV);
  xl_cast_bf16<<<2048, 256, 0, stream>>>(relative, rbf, nKV);
  xl_cast_bf16<<<1024, 256, 0, stream>>>(Wq, wqbf, nW);
  xl_cast_bf16<<<1024, 256, 0, stream>>>(Wk, wkbf, nW);
  xl_cast_bf16<<<1024, 256, 0, stream>>>(Wv, wvbf, nW);
  xl_cast_bf16<<<1024, 256, 0, stream>>>(Wr, wrbf, nW);
  xl_cast_bf16<<<1024, 256, 0, stream>>>(Wo, wobf, nW);

  // 2) projections (WMMA GEMMs)
  dim3 blk(256);
  dim3 gq(DPn / 128, (Bn * TQn) / 128);   // (8, 32)
  dim3 gk(DPn / 128, (Bn * TKn) / 128);   // (8, 48)
  xl_gemm_bf16<0><<<gq, blk, 0, stream>>>(qbf, wqbf, Bn * TQn, DPn, Dn,
                                          nullptr, nullptr, quB, u, qvB, v);
  xl_gemm_bf16<1><<<gk, blk, 0, stream>>>(kvbf, wkbf, Bn * TKn, DPn, Dn,
                                          nullptr, nullptr, kpB, nullptr, nullptr, nullptr);
  xl_gemm_bf16<1><<<gk, blk, 0, stream>>>(kvbf, wvbf, Bn * TKn, DPn, Dn,
                                          nullptr, nullptr, vpB, nullptr, nullptr, nullptr);
  xl_gemm_bf16<1><<<gk, blk, 0, stream>>>(rbf, wrbf, Bn * TKn, DPn, Dn,
                                          nullptr, nullptr, rpB, nullptr, nullptr, nullptr);

  // 3) flash attention (WMMA scores + WMMA ctx)
  dim3 ga(Bn * Hn, TQn / TQB);            // (64, 16)
  xl_attention<<<ga, dim3(128), 0, stream>>>(quB, qvB, kpB, rpB, vpB, ctxB,
                                             Bn, TQn, TKn, Hn);

  // 4) output projection + residual
  xl_gemm_bf16<2><<<gq, blk, 0, stream>>>(ctxB, wobf, Bn * TQn, Dn, DPn,
                                          preln, query, nullptr, nullptr, nullptr, nullptr);

  // 5) layernorm
  xl_layernorm<<<dim3(Bn * TQn), dim3(256), 0, stream>>>(preln, gm, bt, out, Dn);
}
